// CausalSelfAttention_7378753814610
// MI455X (gfx1250) — matrix operand order repaired and checked by reading, not compiled, not run on hardware
//
#include <hip/hip_runtime.h>

typedef __bf16 bf16;
typedef __attribute__((ext_vector_type(16))) __bf16 v16bf;
typedef __attribute__((ext_vector_type(8)))  __bf16 v8bf;
typedef __attribute__((ext_vector_type(8)))  float   v8f;

typedef unsigned int u32x4 __attribute__((ext_vector_type(4)));
typedef int          i32x4 __attribute__((ext_vector_type(4)));
typedef int          i32x8 __attribute__((ext_vector_type(8)));

#define N_EMBD   2048
#define N_HEAD   16
#define HEAD_DIM 128
#define SEQ_T    2048

#if __has_include(<hip/amd_detail/amd_gfx1250_TDM.h>)
#define TDM_6ARG 1     // therock-10.0 headers -> clang-23 6-arg builtin
#else
#define TDM_6ARG 0     // ROCm 7.2 clang-22 -> 5-arg builtin
#endif

static __device__ __forceinline__ v16bf mk16(v8bf lo, v8bf hi){
  v16bf r;
#pragma unroll
  for(int i=0;i<8;i++){ r[i]=lo[i]; r[i+8]=hi[i]; }
  return r;
}

static __device__ __forceinline__ v8f v8f_zero(){
  v8f r;
#pragma unroll
  for(int i=0;i<8;i++) r[i]=0.0f;
  return r;
}

static __device__ __forceinline__ v8f wmma_bf16(v16bf a, v16bf b, v8f c){
  // D = A(16x32 bf16) * B(32x16 bf16) + C(16x16 f32)
  return __builtin_amdgcn_wmma_f32_16x16x32_bf16(false, a, false, b, (short)0, c, false, false);
}

// ---------------------------------------------------------------------------
// TDM: async DMA of a 2D bf16 tile (tile_d0 x tile_d1 elements, row stride
// stride_elems in the tensor) from global memory into LDS at lds_addr.
// Optional LDS padding: pad_interval/pad_amount codes per D# group1 spec.
// Issue from ONE wave; complete with s_wait_tensorcnt 0 + barrier.
// ---------------------------------------------------------------------------
static __device__ __forceinline__ void tdm_load_2d_bf16(
    unsigned lds_addr, const void* gaddr,
    unsigned tile_d0, unsigned tile_d1,
    unsigned tensor_d0, unsigned tensor_d1, unsigned stride_elems,
    unsigned pad_interval_code, unsigned pad_amount_code, bool pad_en)
{
  const unsigned long long ga = (unsigned long long)gaddr;
  u32x4 g0;
  g0[0] = 1u;                                            // count=1 (valid user D#)
  g0[1] = lds_addr;                                      // LDS byte address
  g0[2] = (unsigned)(ga & 0xffffffffu);                  // global_addr[31:0]
  g0[3] = (unsigned)((ga >> 32) & 0x01ffffffu)           // global_addr[56:32]
        | (2u << 30);                                    // type = 2 ("image")
  i32x8 g1;
  unsigned w0 = (1u << 16);                              // data_size = 1 -> 2 bytes
  if (pad_en) w0 |= (1u << 20)                           // pad_enable
            | (pad_interval_code << 22)                  // DWORD interval code
            | (pad_amount_code   << 25);                 // DWORD amount code
  g1[0] = (int)w0;
  g1[1] = (int)((tensor_d0 & 0xffffu) << 16);            // tensor_dim0[15:0]
  g1[2] = (int)(((tensor_d0 >> 16) & 0xffffu)            // tensor_dim0[31:16]
        |        ((tensor_d1 & 0xffffu) << 16));         // tensor_dim1[15:0]
  g1[3] = (int)(((tensor_d1 >> 16) & 0xffffu)            // tensor_dim1[31:16]
        |        ((tile_d0 & 0xffffu) << 16));           // tile_dim0
  g1[4] = (int)(tile_d1 & 0xffffu);                      // tile_dim1 (tile_dim2=0)
  g1[5] = (int)stride_elems;                             // tensor_dim0_stride[31:0]
  g1[6] = 0;                                             // stride[47:32], dim1_stride lo
  g1[7] = 0;
  const i32x4 z4 = {0,0,0,0};
#if TDM_6ARG
  const i32x8 z8 = {0,0,0,0,0,0,0,0};
  __builtin_amdgcn_tensor_load_to_lds(g0, g1, z4, z4, z8, 0);
#else
  __builtin_amdgcn_tensor_load_to_lds(g0, g1, z4, z4, 0);
#endif
}

// ---------------------------------------------------------------------------
// fp32 -> bf16 elementwise convert (n must be a multiple of 4)
// ---------------------------------------------------------------------------
__global__ __launch_bounds__(256) void cvt_f32_bf16(const float* __restrict__ in,
                                                    bf16* __restrict__ out, long n){
  long i = ((long)blockIdx.x*256 + threadIdx.x)*4;
  long stride = (long)gridDim.x*1024;
  for(; i+3 < n; i += stride){
    float4 v = *(const float4*)(in + i);
    out[i+0]=(bf16)v.x; out[i+1]=(bf16)v.y; out[i+2]=(bf16)v.z; out[i+3]=(bf16)v.w;
  }
}

// ---------------------------------------------------------------------------
// C[M,N](f32 or bf16) = A[M,K](bf16, row-major) * B[N,K](bf16, row-major)^T
// 128x128 block tile, K-step 32, 8 waves (wave32) of 64x32 each.
// Tile movement: Tensor Data Mover (double-buffered LDS, TENSORcnt).
// LDS row stride 56 bf16 (112B) produced by TDM padding:
//   row = 64B = 16 DWORDs  -> pad_interval code 3
//   pad = 48B = 12 DWORDs  -> pad_amount  code 11
// ---------------------------------------------------------------------------
#define GS 56

template<bool OUT_BF16>
__global__ __launch_bounds__(256) void gemm_bf16_tn(
    const bf16* __restrict__ A, const bf16* __restrict__ B, void* __restrict__ Cout,
    int M, int N, int K)
{
  __shared__ __align__(16) bf16 As[2][128*GS];
  __shared__ __align__(16) bf16 Bs[2][128*GS];
  const int tid  = threadIdx.x;
  const int wid  = tid >> 5, lane = tid & 31;
  const int lo16 = lane & 15, hi = lane >> 4;
  const int wm = wid & 1, wn = wid >> 1;             // 2x4 wave grid
  const int mBase = blockIdx.y*128, nBase = blockIdx.x*128;

  const unsigned asLds = (unsigned)(size_t)&As[0][0];   // LDS byte offsets
  const unsigned bsLds = (unsigned)(size_t)&Bs[0][0];
  const unsigned bufBytes = 128*GS*2;
  const bf16* Atile = A + (size_t)mBase*K;
  const bf16* Btile = B + (size_t)nBase*K;

  v8f acc[4][2];
#pragma unroll
  for(int i=0;i<4;i++)
#pragma unroll
    for(int j=0;j<2;j++) acc[i][j] = v8f_zero();

  const int nk = K/32;

  // prologue: DMA k-step 0 into buffer 0
  if (wid == 0){
    tdm_load_2d_bf16(asLds, Atile, 32, 128, (unsigned)K, (unsigned)M,
                     (unsigned)K, 3u, 11u, true);
    tdm_load_2d_bf16(bsLds, Btile, 32, 128, (unsigned)K, (unsigned)N,
                     (unsigned)K, 3u, 11u, true);
    __builtin_amdgcn_s_wait_tensorcnt(0);
  }
  __syncthreads();

  for(int kt=0; kt<nk; ++kt){
    const int cur = kt & 1;
    if (wid == 0 && kt+1 < nk){
      const unsigned nb = (unsigned)(cur^1) * bufBytes;
      const size_t ko = (size_t)(kt+1)*32;
      tdm_load_2d_bf16(asLds + nb, Atile + ko, 32, 128, (unsigned)K, (unsigned)M,
                       (unsigned)K, 3u, 11u, true);
      tdm_load_2d_bf16(bsLds + nb, Btile + ko, 32, 128, (unsigned)K, (unsigned)N,
                       (unsigned)K, 3u, 11u, true);
    }

    v16bf af[4], bfg[2];
#pragma unroll
    for(int i=0;i<4;i++){
      // A-frag: lane (lo16) = row, lane-half selects K {0..7,16..23} vs {8..15,24..31}
      const bf16* p = &As[cur][(wm*64 + i*16 + lo16)*GS + hi*8];
      af[i] = mk16(*(const v8bf*)p, *(const v8bf*)(p+16));
    }
#pragma unroll
    for(int j=0;j<2;j++){
      // B-frag: lane = column N, lane-half selects K {0..7,16..23} vs {8..15,24..31}
      const bf16* p = &Bs[cur][(wn*32 + j*16 + lo16)*GS + hi*8];
      bfg[j] = mk16(*(const v8bf*)p, *(const v8bf*)(p+16));
    }
#pragma unroll
    for(int i=0;i<4;i++)
#pragma unroll
      for(int j=0;j<2;j++)
        acc[i][j] = wmma_bf16(af[i], bfg[j], acc[i][j]);

    if (wid == 0) __builtin_amdgcn_s_wait_tensorcnt(0);
    __syncthreads();
  }

#pragma unroll
  for(int i=0;i<4;i++)
#pragma unroll
    for(int j=0;j<2;j++)
#pragma unroll
      for(int e=0;e<8;e++){
        const int row = mBase + wm*64 + i*16 + hi*8 + e;  // C/D layout: M = vgpr + 8*hi
        const int col = nBase + wn*32 + j*16 + lo16;      // N = lane&15
        const float v = acc[i][j][e];
        if (OUT_BF16) ((bf16*)Cout)[(size_t)row*N + col] = (bf16)v;
        else          ((float*)Cout)[(size_t)row*N + col] = v;
      }
}

// ---------------------------------------------------------------------------
// Fused RMSNorm(g) + RoPE on q,k; scatter q,k,v into [B,H,T,hd] bf16.
// One block per token, 256 threads x 8 channels.
// ---------------------------------------------------------------------------
__global__ __launch_bounds__(256) void rmsnorm_rope_kernel(
    const bf16* __restrict__ qkv, const float* __restrict__ g,
    bf16* __restrict__ Q, bf16* __restrict__ K, bf16* __restrict__ V)
{
  __shared__ float red[8];
  const int tok = blockIdx.x;
  const int b = tok >> 11, t = tok & 2047;
  const int tid = threadIdx.x;
  const bf16* row = qkv + (size_t)tok * 3 * N_EMBD;
  const int c0 = tid * 8;
  const int h = c0 >> 7, dbase = c0 & 127;

  float gv[8];
#pragma unroll
  for(int e=0;e<8;e++) gv[e] = g[c0+e];

  for(int sel=0; sel<2; ++sel){
    const bf16* p = row + sel*N_EMBD + c0;
    float x[8]; float ss = 0.f;
#pragma unroll
    for(int e=0;e<8;e++){ x[e] = (float)p[e]; ss += x[e]*x[e]; }
#pragma unroll
    for(int off=16; off>=1; off>>=1) ss += __shfl_xor(ss, off, 32);
    if ((tid & 31) == 0) red[tid>>5] = ss;
    __syncthreads();
    float tot = 0.f;
#pragma unroll
    for(int w=0; w<8; ++w) tot += red[w];
    const float rstd = rsqrtf(tot * (1.0f/2048.0f) + 1e-6f);

    bf16* dst = (sel ? K : Q) + ((size_t)(b*N_HEAD + h)*SEQ_T + t)*HEAD_DIM + dbase;
#pragma unroll
    for(int i=0;i<4;++i){
      const int d0 = dbase + 2*i, d1 = d0 + 1;
      const float n0 = x[2*i]   * rstd * gv[2*i];
      const float n1 = x[2*i+1] * rstd * gv[2*i+1];
      // pos[j] = 10000^(-(4j+1)/128), ang[t,d] = t*pos[d%64]
      const float f0 = __powf(10000.f, -(4.0f*(float)(d0 & 63) + 1.0f) * (1.0f/128.0f));
      const float f1 = __powf(10000.f, -(4.0f*(float)(d1 & 63) + 1.0f) * (1.0f/128.0f));
      const float a0 = (float)t * f0, a1 = (float)t * f1;
      // rotate pairs: out[2j] = x0*cos - x1*sin ; out[2j+1] = x1*cos + x0*sin
      dst[2*i]   = (bf16)(n0*__cosf(a0) - n1*__sinf(a0));
      dst[2*i+1] = (bf16)(n1*__cosf(a1) + n0*__sinf(a1));
    }
    __syncthreads();
  }

  const bf16* pv = row + 2*N_EMBD + c0;
  bf16* dv = V + ((size_t)(b*N_HEAD + h)*SEQ_T + t)*HEAD_DIM + dbase;
#pragma unroll
  for(int e=0;e<8;e++) dv[e] = pv[e];
}

// ---------------------------------------------------------------------------
// Flash attention: grid (T/64, B*H), 128 threads (4 waves x 16 q-rows).
// K tile [64 x 128] row-major in LDS; V tile transposed [128 x 64];
// per-wave LDS staging for P (C-layout -> A-layout).
// ---------------------------------------------------------------------------
__global__ __launch_bounds__(128) void flash_attn_kernel(
    const bf16* __restrict__ Q, const bf16* __restrict__ K,
    const bf16* __restrict__ V, bf16* __restrict__ Y)
{
  __shared__ __align__(16) bf16 Ks[64*136];    // stride 136 (272B, bank-spread)
  __shared__ __align__(16) bf16 Vt[128*72];    // stride 72  (144B)
  __shared__ __align__(16) bf16 Ps[4][16*72];

  const int tid  = threadIdx.x;
  const int wid  = tid >> 5, lane = tid & 31;
  const int lo16 = lane & 15, hi = lane >> 4;
  const int bh = blockIdx.y;
  const int b = bh >> 4, h = bh & 15;
  const int q0 = blockIdx.x * 64;
  const float scale = 0.088388347648318447f;   // 1/sqrt(128)

  const bf16* qptr = Q + (size_t)bh * SEQ_T * HEAD_DIM;
  const bf16* kptr = K + (size_t)bh * SEQ_T * HEAD_DIM;
  const bf16* vptr = V + (size_t)bh * SEQ_T * HEAD_DIM;

  // Q fragments (16 rows x 128) held in registers for the whole sweep
  v16bf qf[4];
  {
    const bf16* qr = qptr + (size_t)(q0 + wid*16 + lo16) * HEAD_DIM;
#pragma unroll
    for(int kk=0; kk<4; ++kk){
      v8bf lo = *(const v8bf*)(qr + kk*32 + hi*8);
      v8bf hp = *(const v8bf*)(qr + kk*32 + 16 + hi*8);
      qf[kk] = mk16(lo, hp);
    }
  }

  float m_e[8], l_e[8];
  v8f acc[8];
#pragma unroll
  for(int e=0;e<8;e++){ m_e[e] = -3.0e38f; l_e[e] = 0.f; }
#pragma unroll
  for(int o=0;o<8;o++) acc[o] = v8f_zero();

  for(int j0=0; j0<=q0; j0+=64){
    __syncthreads();
    // K tile -> LDS (row-major)
#pragma unroll
    for(int vi=0; vi<8; ++vi){
      const int slot = vi*128 + tid;
      const int row = slot >> 4, cc = slot & 15;
      *(v8bf*)(&Ks[row*136 + cc*8]) =
          *(const v8bf*)(kptr + (size_t)(j0+row)*HEAD_DIM + cc*8);
    }
    // V tile -> LDS, transposed to [hd][key]
#pragma unroll
    for(int vi=0; vi<8; ++vi){
      const int slot = vi*128 + tid;
      const int row = slot >> 4, cc = slot & 15;
      v8bf vv = *(const v8bf*)(vptr + (size_t)(j0+row)*HEAD_DIM + cc*8);
#pragma unroll
      for(int e=0;e<8;e++) Vt[(cc*8+e)*72 + row] = vv[e];
    }
    __syncthreads();

    // S = Q * K^T  (16 x 64), scaled + causal-masked
    v8f s[4];
#pragma unroll
    for(int n=0;n<4;++n){
      v8f sv = v8f_zero();
#pragma unroll
      for(int kk=0; kk<4; ++kk){
        const bf16* p = &Ks[(n*16 + lo16)*136 + kk*32 + hi*8];
        v16bf bfrag = mk16(*(const v8bf*)p, *(const v8bf*)(p+16));
        sv = wmma_bf16(qf[kk], bfrag, sv);
      }
      const int colg = j0 + n*16 + lo16;
#pragma unroll
      for(int e=0;e<8;e++){
        const int rowg = q0 + wid*16 + hi*8 + e;
        const float xx = sv[e] * scale;
        sv[e] = (colg <= rowg) ? xx : -3.0e38f;
      }
      s[n] = sv;
    }

    // online softmax: row reductions live in a 16-lane half-wave
    float alpha[8];
#pragma unroll
    for(int e=0;e<8;e++){
      float v = fmaxf(fmaxf(s[0][e], s[1][e]), fmaxf(s[2][e], s[3][e]));
      v = fmaxf(v, __shfl_xor(v, 1, 32));
      v = fmaxf(v, __shfl_xor(v, 2, 32));
      v = fmaxf(v, __shfl_xor(v, 4, 32));
      v = fmaxf(v, __shfl_xor(v, 8, 32));
      const float mnew = fmaxf(m_e[e], v);
      alpha[e] = __expf(m_e[e] - mnew);
      m_e[e] = mnew;
    }
#pragma unroll
    for(int n=0;n<4;++n)
#pragma unroll
      for(int e=0;e<8;e++)
        s[n][e] = __expf(s[n][e] - m_e[e]);
#pragma unroll
    for(int e=0;e<8;e++){
      float rs = s[0][e]+s[1][e]+s[2][e]+s[3][e];
      rs += __shfl_xor(rs, 1, 32);
      rs += __shfl_xor(rs, 2, 32);
      rs += __shfl_xor(rs, 4, 32);
      rs += __shfl_xor(rs, 8, 32);
      l_e[e] = l_e[e]*alpha[e] + rs;
    }
#pragma unroll
    for(int o=0;o<8;++o)
#pragma unroll
      for(int e=0;e<8;e++)
        acc[o][e] = acc[o][e]*alpha[e];

    // P (C-layout) -> per-wave LDS -> A-layout fragments
#pragma unroll
    for(int n=0;n<4;++n)
#pragma unroll
      for(int e=0;e<8;e++)
        Ps[wid][(hi*8+e)*72 + n*16 + lo16] = (bf16)s[n][e];
    asm volatile("s_wait_dscnt 0x0" ::: "memory");

    v16bf pf[2];
#pragma unroll
    for(int kk=0; kk<2; ++kk){
      const bf16* p = &Ps[wid][lo16*72 + kk*32 + hi*8];
      pf[kk] = mk16(*(const v8bf*)p, *(const v8bf*)(p+16));
    }
    // O += P * V   (K dim = 64 keys = 2 wmma k-steps, 8 hd tiles)
#pragma unroll
    for(int o=0;o<8;++o){
#pragma unroll
      for(int kk=0;kk<2;++kk){
        const bf16* p = &Vt[(o*16 + lo16)*72 + kk*32 + hi*8];
        v16bf vfrag = mk16(*(const v8bf*)p, *(const v8bf*)(p+16));
        acc[o] = wmma_bf16(pf[kk], vfrag, acc[o]);
      }
    }
  }

  // y = acc / l, scatter back to [B, T, C] (token-major for the projection GEMM)
#pragma unroll
  for(int o=0;o<8;++o)
#pragma unroll
    for(int e=0;e<8;e++){
      const int t = q0 + wid*16 + hi*8 + e;
      const int d = o*16 + lo16;
      const float yv = acc[o][e] / l_e[e];
      Y[((size_t)b*SEQ_T + t)*N_EMBD + h*HEAD_DIM + d] = (bf16)yv;
    }
}

// ---------------------------------------------------------------------------
extern "C" void kernel_launch(void* const* d_in, const int* in_sizes, int n_in,
                              void* d_out, int out_size, void* d_ws, size_t ws_size,
                              hipStream_t stream)
{
  (void)in_sizes; (void)n_in; (void)out_size; (void)ws_size;
  const float* x     = (const float*)d_in[0];
  const float* Wqkv  = (const float*)d_in[1];
  const float* Wproj = (const float*)d_in[2];
  const float* gqk   = (const float*)d_in[3];

  const int Bsz = 2, C = N_EMBD;
  const size_t MT = (size_t)Bsz * SEQ_T;   // 4096 tokens

  char* ws = (char*)d_ws;
  size_t off = 0;
  auto wsalloc = [&](size_t bytes)->char*{
    char* p = ws + off; off += (bytes + 255) & ~(size_t)255; return p;
  };
  bf16* xb     = (bf16*)wsalloc(MT*C*2);                // 16 MB
  bf16* wqkvb  = (bf16*)wsalloc((size_t)3*C*C*2);       // 24 MB
  bf16* wprojb = (bf16*)wsalloc((size_t)C*C*2);         //  8 MB
  bf16* qkvb   = (bf16*)wsalloc(MT*3*C*2);              // 48 MB
  bf16* Qb     = (bf16*)wsalloc(MT*C*2);
  bf16* Kb     = (bf16*)wsalloc(MT*C*2);
  bf16* Vb     = (bf16*)wsalloc(MT*C*2);
  bf16* Yb     = (bf16*)wsalloc(MT*C*2);

  cvt_f32_bf16<<<2048, 256, 0, stream>>>(x,     xb,     (long)(MT*C));
  cvt_f32_bf16<<<2048, 256, 0, stream>>>(Wqkv,  wqkvb,  (long)((size_t)3*C*C));
  cvt_f32_bf16<<<1024, 256, 0, stream>>>(Wproj, wprojb, (long)((size_t)C*C));

  // qkv = x @ Wqkv^T  : [4096,2048] x [6144,2048]^T
  gemm_bf16_tn<true><<<dim3(48, 32), 256, 0, stream>>>(xb, wqkvb, qkvb,
                                                       (int)MT, 3*C, C);
  rmsnorm_rope_kernel<<<(int)MT, 256, 0, stream>>>(qkvb, gqk, Qb, Kb, Vb);

  flash_attn_kernel<<<dim3(SEQ_T/64, Bsz*N_HEAD), 128, 0, stream>>>(Qb, Kb, Vb, Yb);

  // out = y @ Wproj^T : [4096,2048] x [2048,2048]^T  (fp32 output)
  gemm_bf16_tn<false><<<dim3(16, 32), 256, 0, stream>>>(Yb, wprojb, d_out,
                                                        (int)MT, C, C);
}
